// YOLOLayer_11519102288648
// MI455X (gfx1250) — matrix operand order repaired and checked by reading, not compiled, not run on hardware
//
#include <hip/hip_runtime.h>

// ---------------------------------------------------------------------------
// YOLO layer forward (loss) for MI455X / gfx1250, wave32 + WMMA f16->f32.
//
// Pipeline:
//   k0: pack conv_w (255x256 f32) -> padded 256x256 f16 in ws
//   k1: per-label preprocessing (16x50)
//   k2: 1x1-conv GEMM via v_wmma_f32_16x16x32_f16, LDS-staged tiles
//       (W tile staged with global_load_async_to_lds_b128 when available)
//   k3: per-cell IoU + BCE/MSE losses, reduced to 5 scalars via f32 atomics
// ---------------------------------------------------------------------------

typedef __attribute__((ext_vector_type(16))) _Float16 v16h;
typedef __attribute__((ext_vector_type(2)))  _Float16 h16x2;
typedef __attribute__((ext_vector_type(8)))  float    v8f;
typedef __attribute__((ext_vector_type(4)))  float    f32x4;
typedef __attribute__((ext_vector_type(4)))  unsigned int u32x4;

union Frag32B { v16h h; u32x4 u[2]; };

#define BATCH   16
#define CIN     256
#define FS      64
#define HW      4096           // FS*FS
#define NANCH   3
#define NCH     85             // 5 + 80
#define OC      255            // NCH*NANCH
#define OPAD    256
#define KLAB    50

#define PXT     64             // pixels per workgroup in GEMM
#define WROW    528            // bytes per LDS row: 256 f16 (512B) + 16B pad
#define SMEM_W  (256 * WROW)   // W tile LDS bytes
#define SMEM_X  (PXT * WROW)   // X tile LDS bytes
#define SMEM_TOT (SMEM_W + SMEM_X)   // 168960 B, 1 WG per 320KB WGP

// ws layout
#define WS_WH    0                       // 256*256 f16 = 131072 B
#define WS_CONV  131072                  // 16*256*4096 f32 = 67108864 B
#define WS_LAB   (131072 + 67108864)     // 16*50*16 f32 = 51200 B

// Async global->LDS (CDNA5): signature (per compiler diagnostic) is
//   (int4 addrspace(1)* gsrc, int4 addrspace(3)* ldst, imm int off, imm int cpol)
#if __has_builtin(__builtin_amdgcn_global_load_async_to_lds_b128)
#define HAVE_ASYNC_LDS 1
#if __has_builtin(__builtin_amdgcn_s_wait_asynccnt)
#define WAIT_ASYNC0() __builtin_amdgcn_s_wait_asynccnt(0)
#else
#define WAIT_ASYNC0() asm volatile("s_wait_asynccnt 0" ::: "memory")
#endif
typedef int i32x4 __attribute__((vector_size(16)));
typedef __attribute__((address_space(1))) i32x4 as1_i32x4;
typedef __attribute__((address_space(3))) i32x4 as3_i32x4;
#else
#define HAVE_ASYNC_LDS 0
#endif

// REF_ANCH = ANCHORS / 8
__device__ __constant__ float c_refw[9] = {1.25f, 2.0f, 4.125f, 3.75f, 7.75f, 7.375f, 14.5f, 19.5f, 46.625f};
__device__ __constant__ float c_refh[9] = {1.625f, 3.75f, 2.875f, 7.625f, 5.625f, 14.875f, 11.25f, 24.75f, 40.75f};

__device__ __forceinline__ float sigf(float x) { return 1.0f / (1.0f + __expf(-x)); }

__device__ __forceinline__ float bcef(float p, float t) {
    float pc = fminf(fmaxf(p, 1e-12f), 1.0f - 1e-12f);
    float l1 = fmaxf(__logf(pc), -100.0f);
    float l2 = fmaxf(log1pf(-pc), -100.0f);
    return -(t * l1 + (1.0f - t) * l2);
}

// ---------------------------------------------------------------------------
// k0: pack W f32[255][256] -> f16[256][256] (row 255 zero)
// ---------------------------------------------------------------------------
__global__ void pack_w_kernel(const float* __restrict__ w, _Float16* __restrict__ wh) {
    int idx = blockIdx.x * 256 + threadIdx.x;      // 65536 threads
    int o = idx >> 8;
    int c = idx & 255;
    float v = (o < OC) ? w[o * CIN + c] : 0.0f;
    wh[idx] = (_Float16)v;
    (void)c;
}

// ---------------------------------------------------------------------------
// k1: per-label preprocessing.  labinfo[b][k][16]:
//  0..3: tx,ty,tw,th  4: ti  5: tj  6: match  7: a_idx
//  8: fx  9: fy  10: lw  11: lh  12: scale  13: cls  14: valid  15: pad
// ---------------------------------------------------------------------------
__global__ void prep_labels_kernel(const float* __restrict__ labels, float* __restrict__ labinfo) {
    int idx = blockIdx.x * 256 + threadIdx.x;
    if (idx >= BATCH * KLAB) return;
    const float* L = labels + (size_t)idx * 5;
    float cls = L[0], x = L[1], y = L[2], w = L[3], h = L[4];
    float tx = x * (float)FS, ty = y * (float)FS;
    float tw = w * (float)FS, th = h * (float)FS;
    int ti = (int)tx;
    int tj = (int)ty;
    // anchor IoU argmax (first max wins, like jnp.argmax)
    float best = -1.0f; int bestn = 0;
    #pragma unroll
    for (int n = 0; n < 9; ++n) {
        float inter = fminf(tw, c_refw[n]) * fminf(th, c_refh[n]);
        float uni   = tw * th + c_refw[n] * c_refh[n] - inter;
        float iou   = inter / uni;
        if (iou > best) { best = iou; bestn = n; }
    }
    int a = bestn % 3;
    int valid = (cls + x + y + w + h) > 0.0f;
    int match = valid && (bestn < 3);
    float scale = 2.0f - tw * th * (1.0f / (float)(FS * FS));
    float fx = tx - (float)ti;
    float fy = ty - (float)tj;
    float lw = __logf(tw / c_refw[a] + 1e-16f);
    float lh = __logf(th / c_refh[a] + 1e-16f);
    float* o = labinfo + (size_t)idx * 16;
    o[0] = tx;  o[1] = ty;  o[2] = tw;  o[3] = th;
    o[4] = (float)ti; o[5] = (float)tj; o[6] = match ? 1.0f : 0.0f; o[7] = (float)a;
    o[8] = fx;  o[9] = fy;  o[10] = lw; o[11] = lh;
    o[12] = scale; o[13] = cls; o[14] = valid ? 1.0f : 0.0f; o[15] = 0.0f;
}

// ---------------------------------------------------------------------------
// k2: GEMM  conv[b][o][p] = sum_c W[o][c] * X[b][c][p] + bias[o]
// One workgroup: one (batch, 64-pixel tile); all 256 (padded) output rows.
// 8 waves, each owns 8 Mtiles x 1 Ntile of 16x16 accumulators.
// ---------------------------------------------------------------------------
__global__ void __launch_bounds__(256, 2)
gemm_wmma_kernel(const float* __restrict__ xin, const _Float16* __restrict__ wh,
                 const float* __restrict__ bias, float* __restrict__ conv) {
    extern __shared__ char smem[];
    char* Wl = smem;            // [256][WROW]
    char* Xl = smem + SMEM_W;   // [64][WROW], transposed: row = pixel, col = channel

    const int tid = threadIdx.x;
    const int wg  = blockIdx.x;
    const int b   = wg >> 6;          // 64 pixel tiles per batch
    const int pt  = wg & 63;

    // --- stage W (f16, pre-packed) global -> LDS, 16B chunks per lane ---
    {
        int r0  = tid >> 5;                    // 0..7
        int col = tid & 31;                    // 16B chunk within row
#if HAVE_ASYNC_LDS
        #pragma unroll
        for (int rr = 0; rr < 32; ++rr) {
            int r = r0 + rr * 8;
            i32x4* gp = (i32x4*)((char*)wh + ((size_t)r * 32 + col) * 16);
            i32x4* lp = (i32x4*)(Wl + (size_t)r * WROW + col * 16);
            __builtin_amdgcn_global_load_async_to_lds_b128(
                (as1_i32x4*)gp, (as3_i32x4*)lp, 0, 0);
        }
#else
        const u32x4* src = (const u32x4*)wh;   // 8192 x 16B
        #pragma unroll
        for (int rr = 0; rr < 32; ++rr) {
            int r = r0 + rr * 8;
            u32x4 d = src[r * 32 + col];
            *(u32x4*)(Wl + (size_t)r * WROW + col * 16) = d;
        }
#endif
    }
    // --- stage X tile, convert f32 -> f16, transposed packed store Xl[p][c] ---
    {
        int col4 = (tid & 15) * 4;        // pixel quad base
        int cp   = (tid >> 4) * 2;        // channel pair within a 32-channel pass
        const float* xb = xin + (size_t)b * CIN * HW + (size_t)pt * PXT;
        #pragma unroll
        for (int pass = 0; pass < 8; ++pass) {
            int c = pass * 32 + cp;
            f32x4 va = *(const f32x4*)(xb + (size_t)c * HW + col4);
            f32x4 vb = *(const f32x4*)(xb + (size_t)(c + 1) * HW + col4);
            #pragma unroll
            for (int jj = 0; jj < 4; ++jj) {
                h16x2 pk;
                pk.x = (_Float16)va[jj];
                pk.y = (_Float16)vb[jj];
                *(h16x2*)(Xl + (size_t)(col4 + jj) * WROW + c * 2) = pk;
            }
        }
    }
#if HAVE_ASYNC_LDS
    WAIT_ASYNC0();
#endif
    __syncthreads();

    const int wave  = tid >> 5;
    const int lane  = tid & 31;
    const int ln    = lane & 15;
    const int hh    = lane >> 4;      // half-wave select
    const int ntile = wave & 3;       // pixel group (16)
    const int mbase = (wave >> 2) * 8; // 8 M-tiles per wave

    v8f acc[8];
    #pragma unroll
    for (int m = 0; m < 8; ++m)
        #pragma unroll
        for (int e = 0; e < 8; ++e) acc[m][e] = 0.0f;

    const char* Brow = Xl + (size_t)(ntile * 16 + ln) * WROW;

    #pragma unroll
    for (int k = 0; k < 8; ++k) {                // K = 8 * 32 = 256
        // B fragment: lane half h holds K = 8h..8h+7 and 16+8h..16+8h+7
        Frag32B fb;
        const char* pB = Brow + (k * 32 + hh * 8) * 2;
        fb.u[0] = *(const u32x4*)(pB);
        fb.u[1] = *(const u32x4*)(pB + 32);
        #pragma unroll
        for (int m = 0; m < 8; ++m) {
            // A fragment: row M = ln; VGPR0-3 = K h*8..h*8+7, VGPR4-7 = +16
            const char* pA = Wl + (size_t)((mbase + m) * 16 + ln) * WROW + (k * 32 + hh * 8) * 2;
            Frag32B fa;
            fa.u[0] = *(const u32x4*)(pA);
            fa.u[1] = *(const u32x4*)(pA + 32);
            acc[m] = __builtin_amdgcn_wmma_f32_16x16x32_f16(
                false, fa.h, false, fb.h, (short)0, acc[m], false, false);
        }
    }

    // --- epilogue: +bias, store raw conv activations ---
    const int p = pt * PXT + ntile * 16 + ln;    // C layout: N = lane&15
    float* outb = conv + (size_t)b * OPAD * HW + p;
    #pragma unroll
    for (int m = 0; m < 8; ++m) {
        int obase = (mbase + m) * 16 + 8 * hh;   // C layout: M = r + 8*(lane>>4)
        #pragma unroll
        for (int r = 0; r < 8; ++r) {
            int o = obase + r;
            if (o < OC) outb[(size_t)o * HW] = acc[m][r] + bias[o];
        }
    }
}

// ---------------------------------------------------------------------------
// k3: per-cell loss.  One thread per (b, a, j, i).
// ---------------------------------------------------------------------------
__global__ void __launch_bounds__(256)
loss_kernel(const float* __restrict__ conv, const float* __restrict__ labinfo,
            float* __restrict__ out5) {
    const int tid = threadIdx.x;
    const int gid = blockIdx.x * 256 + tid;          // 0..196607
    const int b   = gid / (NANCH * HW);              // block-uniform
    const int rem = gid % (NANCH * HW);
    const int a   = rem / HW;
    const int ji  = rem % HW;
    const int j   = ji >> 6;
    const int i   = ji & 63;

    __shared__ float lab[KLAB * 16];
    for (int t = tid; t < KLAB * 16; t += 256) lab[t] = labinfo[(size_t)b * KLAB * 16 + t];
    __syncthreads();

    const float* cbase = conv + ((size_t)b * OPAD + a * NCH) * HW + ji;
    const float v0 = cbase[0];
    const float v1 = cbase[(size_t)HW];
    const float v2 = cbase[(size_t)2 * HW];
    const float v3 = cbase[(size_t)3 * HW];
    const float v4 = cbase[(size_t)4 * HW];

    // pred box
    const float px = sigf(v0) + (float)i;
    const float py = sigf(v1) + (float)j;
    const float pw = __expf(v2) * c_refw[a];
    const float ph = __expf(v3) * c_refh[a];
    const float axl = px - pw * 0.5f, ayl = py - ph * 0.5f;
    const float axr = px + pw * 0.5f, ayr = py + ph * 0.5f;
    const float area_a = pw * ph;

    float best_iou = 0.0f;
    int kmatch = -1;
    #pragma unroll 5
    for (int k = 0; k < KLAB; ++k) {
        const float* L = lab + k * 16;
        float tx = L[0], ty = L[1], tw = L[2], th = L[3];
        float bxl = tx - tw * 0.5f, byl = ty - th * 0.5f;
        float bxr = tx + tw * 0.5f, byr = ty + th * 0.5f;
        float iw = fminf(axr, bxr) - fmaxf(axl, bxl);
        float ih = fminf(ayr, byr) - fmaxf(ayl, byl);
        iw = fmaxf(iw, 0.0f); ih = fmaxf(ih, 0.0f);
        float inter = iw * ih;
        float iou = inter / (area_a + tw * th - inter);
        best_iou = fmaxf(best_iou, iou);
        if (L[6] != 0.0f && (int)L[7] == a && (int)L[4] == i && (int)L[5] == j)
            kmatch = k;       // last matching label wins (scatter semantics)
    }

    float tm = 0.0f, tscale = 0.0f, fx = 0.0f, fy = 0.0f, lw = 0.0f, lh = 0.0f, obj;
    int cls = -1;
    if (kmatch >= 0) {
        const float* L = lab + kmatch * 16;
        tm = 1.0f; tscale = L[12];
        fx = L[8]; fy = L[9]; lw = L[10]; lh = L[11];
        cls = (int)L[13];
        obj = 1.0f;
    } else {
        obj = (best_iou > 0.7f) ? 0.0f : 1.0f;
    }

    float lxy = 0.0f, lwh = 0.0f, lobj = 0.0f, lcls = 0.0f;
    lxy += bcef(sigf(v0) * tm, fx * tm) * tscale;
    lxy += bcef(sigf(v1) * tm, fy * tm) * tscale;
    float d2 = (v2 - lw) * tm; lwh += 0.5f * d2 * d2 * tscale;
    float d3 = (v3 - lh) * tm; lwh += 0.5f * d3 * d3 * tscale;
    lobj = bcef(sigf(v4) * obj, tm * obj);
    for (int c = 5; c < NCH; ++c) {
        float v = cbase[(size_t)c * HW];
        float t = ((c - 5) == cls) ? tm : 0.0f;
        lcls += bcef(sigf(v) * tm, t);
    }

    // block reduction of the 4 partials
    __shared__ float r0[256], r1[256], r2[256], r3[256];
    r0[tid] = lxy; r1[tid] = lwh; r2[tid] = lobj; r3[tid] = lcls;
    __syncthreads();
    for (int s = 128; s > 0; s >>= 1) {
        if (tid < s) {
            r0[tid] += r0[tid + s];
            r1[tid] += r1[tid + s];
            r2[tid] += r2[tid + s];
            r3[tid] += r3[tid + s];
        }
        __syncthreads();
    }
    if (tid == 0) {
        atomicAdd(out5 + 1, r0[0]);
        atomicAdd(out5 + 2, r1[0]);
        atomicAdd(out5 + 3, r2[0]);
        atomicAdd(out5 + 4, r3[0]);
        atomicAdd(out5 + 0, r0[0] + r1[0] + r2[0] + r3[0]);
    }
}

// ---------------------------------------------------------------------------
extern "C" void kernel_launch(void* const* d_in, const int* in_sizes, int n_in,
                              void* d_out, int out_size, void* d_ws, size_t ws_size,
                              hipStream_t stream) {
    const float* xin    = (const float*)d_in[0];
    const float* labels = (const float*)d_in[1];
    const float* conv_w = (const float*)d_in[2];
    const float* conv_b = (const float*)d_in[3];
    float* out = (float*)d_out;
    char*  ws  = (char*)d_ws;

    _Float16* wh      = (_Float16*)(ws + WS_WH);
    float*    conv    = (float*)(ws + WS_CONV);
    float*    labinfo = (float*)(ws + WS_LAB);

    (void)hipMemsetAsync(d_out, 0, 5 * sizeof(float), stream);

    pack_w_kernel<<<OPAD * CIN / 256, 256, 0, stream>>>(conv_w, wh);
    prep_labels_kernel<<<(BATCH * KLAB + 255) / 256, 256, 0, stream>>>(labels, labinfo);
    gemm_wmma_kernel<<<BATCH * (HW / PXT), 256, SMEM_TOT, stream>>>(xin, wh, conv_b, conv);
    loss_kernel<<<BATCH * NANCH * HW / 256, 256, 0, stream>>>(conv, labinfo, out);

    (void)in_sizes; (void)n_in; (void)out_size; (void)ws_size;
}
